// GraphConvolutionalLSTM_MIL0_86285892976679
// MI455X (gfx1250) — matrix operand order repaired and checked by reading, not compiled, not run on hardware
//
#include <hip/hip_runtime.h>
#include <math.h>

// ---- shapes ----
#define B_    32
#define T_    48
#define N_    360
#define H_    100
#define D_    64
#define G4    400     // 4*H
#define KPADX 384     // N_ padded to mult of 32
#define KPADH 128     // H_ padded to mult of 32
#define RB    1536    // B_*T_

typedef __attribute__((ext_vector_type(16))) __bf16 v16bf;
typedef __attribute__((ext_vector_type(8)))  __bf16 v8bf;
typedef __attribute__((ext_vector_type(8)))  float  v8f;

// ---------------- WMMA fragment loaders (16x16x32 bf16) ----------------
// A: 16(M) x 32(K). lane L: M = 16*mt + (L&15). VGPR0-3 hold K = k0..k0+7,
// VGPR4-7 hold K = k0+16..k0+23, where k0 = 32*kb + ((L&16)?8:0).
__device__ __forceinline__ v16bf load_frag_a(const __bf16* __restrict__ base,
                                             int ldk, int mt, int kb, int lane) {
  const int m  = mt * 16 + (lane & 15);
  const int k0 = kb * 32 + ((lane & 16) ? 8 : 0);
  const __bf16* p = base + (size_t)m * ldk + k0;
  v8bf lo = *(const v8bf*)(p);
  v8bf hi = *(const v8bf*)(p + 16);
  v16bf a;
#pragma unroll
  for (int i = 0; i < 8; ++i) { a[i] = lo[i]; a[i + 8] = hi[i]; }
  return a;
}

// B: 32(K) x 16(N), weights stored row-major [n][K]. lane L: N = 16*nt + (L&15),
// holds K = k0..k0+7 and k0+16..k0+23, where k0 = 32*kb + ((L&16)?8:0).
__device__ __forceinline__ v16bf load_frag_b(const __bf16* __restrict__ base,
                                             int ldk, int nt, int kb, int lane) {
  const int n  = nt * 16 + (lane & 15);
  const int k0 = kb * 32 + ((lane & 16) ? 8 : 0);
  const __bf16* p = base + (size_t)n * ldk + k0;
  v8bf lo = *(const v8bf*)(p);
  v8bf hi = *(const v8bf*)(p + 16);
  v16bf b;
#pragma unroll
  for (int i = 0; i < 8; ++i) { b[i] = lo[i]; b[i + 8] = hi[i]; }
  return b;
}

// ---------------- Kernel 1: GCN (HBM-bound streaming matvec) ----------------
// gout[r=t*32+b][n] = relu(dot(x[b][t][n][:], w) + bias) -> bf16 xpad[r][KPADX]
__global__ __launch_bounds__(256) void k_gcn(const float* __restrict__ x,
                                             const float* __restrict__ w,
                                             const float* __restrict__ bptr,
                                             __bf16* __restrict__ xpad) {
  const int lane = threadIdx.x & 31;
  const int wid  = blockIdx.x * 8 + (threadIdx.x >> 5);   // one wave per output
  const int r = wid / N_;                                 // 0..1535 (= t*32+b)
  const int n = wid - r * N_;                             // 0..359
  const int b = r & 31;
  const int t = r >> 5;
  const float4* __restrict__ xr = (const float4*)(x + ((size_t)(b * T_ + t) * N_ + n) * N_);
  const float4* __restrict__ w4 = (const float4*)w;       // 360 = 90 float4
  float s = 0.f;
#pragma unroll
  for (int i = 0; i < 3; ++i) {
    int m4 = lane + 32 * i;
    if (m4 < 90) {
      float4 a = xr[m4], c = w4[m4];
      s += a.x * c.x + a.y * c.y + a.z * c.z + a.w * c.w;
    }
  }
#pragma unroll
  for (int off = 16; off > 0; off >>= 1) s += __shfl_down(s, off, 32);
  if (lane == 0) {
    float v = fmaxf(s + bptr[0], 0.f);
    xpad[(size_t)r * KPADX + n] = (__bf16)v;
  }
}

// ---------------- Kernel 2: pack weights to bf16 (+ zero x K-padding) -------
__global__ __launch_bounds__(256) void k_pack(const float* __restrict__ Wih,
                                              const float* __restrict__ Whh,
                                              __bf16* __restrict__ wihb,
                                              __bf16* __restrict__ whhb,
                                              __bf16* __restrict__ xpad) {
  const int idx = blockIdx.x * 256 + threadIdx.x;   // grid covers 400*384
  const int row = idx / KPADX;
  const int k   = idx - row * KPADX;
  if (row < G4) {
    wihb[idx] = (k < N_) ? (__bf16)Wih[row * N_ + k] : (__bf16)0.f;
    if (k < KPADH)
      whhb[row * KPADH + k] = (k < H_) ? (__bf16)Whh[row * H_ + k] : (__bf16)0.f;
  }
  if (idx < RB * (KPADX - N_)) {                    // zero x K-padding
    int rr = idx / (KPADX - N_);
    int kk = N_ + idx - rr * (KPADX - N_);
    xpad[(size_t)rr * KPADX + kk] = (__bf16)0.f;
  }
}

// ---------------- Kernel 3: parallel GEMM  Gx = Xpad @ Wih^T  (WMMA) --------
// M=1536, N=400, K=384. One 16x16 tile per wave, 12 K-blocks of 32.
__global__ __launch_bounds__(256) void k_gemm_x(const __bf16* __restrict__ xpad,
                                                const __bf16* __restrict__ wihb,
                                                float* __restrict__ gx) {
  const int lane = threadIdx.x & 31;
  const int tile = blockIdx.x * 8 + (threadIdx.x >> 5);   // 0..2399
  const int mt = tile / 25;
  const int nt = tile - mt * 25;
  v8f acc = {};
#pragma unroll 4
  for (int kb = 0; kb < 12; ++kb) {
    v16bf a = load_frag_a(xpad, KPADX, mt, kb, lane);
    v16bf b = load_frag_b(wihb, KPADX, nt, kb, lane);
    acc = __builtin_amdgcn_wmma_f32_16x16x32_bf16(false, a, false, b,
                                                  (short)0, acc, false, false);
  }
  const int col   = nt * 16 + (lane & 15);
  const int mbase = mt * 16 + ((lane & 16) ? 8 : 0);
#pragma unroll
  for (int v = 0; v < 8; ++v)
    gx[(size_t)(mbase + v) * G4 + col] = acc[v];
}

// ---------------- Kernel 4: sequential LSTM (recurrent WMMA on one WGP) -----
__global__ __launch_bounds__(512) void k_lstm(const float* __restrict__ gx,
                                              const __bf16* __restrict__ whhb,
                                              const float* __restrict__ bih,
                                              const float* __restrict__ bhh,
                                              float* __restrict__ hs) {
  __shared__ float  gates[32 * G4];                 // 51200 B
  __shared__ alignas(16) __bf16 hbf[32 * KPADH];    //  8192 B
  __shared__ float  sbias[G4];                      //  1600 B
  const int tid  = threadIdx.x;
  const int lane = tid & 31;
  const int wave = tid >> 5;                        // 0..15
  for (int i = tid; i < G4; i += 512) sbias[i] = bih[i] + bhh[i];
  for (int i = tid; i < 32 * KPADH; i += 512) hbf[i] = (__bf16)0.f;
  float creg[7];
#pragma unroll
  for (int r = 0; r < 7; ++r) creg[r] = 0.f;
  __syncthreads();

  for (int t = 0; t < T_; ++t) {
    // gates = Gx[t] + h @ Whh^T   (50 tiles of 16x16 across 16 waves)
    for (int tile = wave; tile < 50; tile += 16) {
      const int mt  = tile & 1;
      const int nt  = tile >> 1;
      const int col = nt * 16 + (lane & 15);
      const int mb  = mt * 16 + ((lane & 16) ? 8 : 0);
      v8f acc;
#pragma unroll
      for (int v = 0; v < 8; ++v)
        acc[v] = gx[(size_t)(t * 32 + mb + v) * G4 + col];   // C init = x-part
#pragma unroll
      for (int kb = 0; kb < 4; ++kb) {
        v16bf a = load_frag_a(hbf, KPADH, mt, kb, lane);     // LDS ds_read
        v16bf b = load_frag_b(whhb, KPADH, nt, kb, lane);
        acc = __builtin_amdgcn_wmma_f32_16x16x32_bf16(false, a, false, b,
                                                      (short)0, acc, false, false);
      }
#pragma unroll
      for (int v = 0; v < 8; ++v)
        gates[(mb + v) * G4 + col] = acc[v];
    }
    __syncthreads();
    // elementwise cell update; c lives in registers (static ownership)
#pragma unroll
    for (int r = 0; r < 7; ++r) {
      int e = tid + 512 * r;
      if (e < 3200) {
        int bb = e / 100, j = e - bb * 100;
        float gi = gates[bb * G4 + j]       + sbias[j];
        float gf = gates[bb * G4 + 100 + j] + sbias[100 + j];
        float gg = gates[bb * G4 + 200 + j] + sbias[200 + j];
        float go = gates[bb * G4 + 300 + j] + sbias[300 + j];
        float i_ = 1.f / (1.f + __expf(-gi));
        float f_ = 1.f / (1.f + __expf(-gf));
        float o_ = 1.f / (1.f + __expf(-go));
        float c  = f_ * creg[r] + i_ * tanhf(gg);
        creg[r]  = c;
        float h  = o_ * tanhf(c);
        hs[(size_t)(t * 32 + bb) * H_ + j] = h;
        hbf[bb * KPADH + j] = (__bf16)h;     // cols 100..127 stay zero
      }
    }
    __syncthreads();
  }
}

// ---------------- Kernel 5: MIL attention + softmax + classifier ------------
__global__ __launch_bounds__(128) void k_attn(const float* __restrict__ hs,
                                              const float* __restrict__ wa1,
                                              const float* __restrict__ ba1,
                                              const float* __restrict__ wa2,
                                              const float* __restrict__ ba2,
                                              const float* __restrict__ wc,
                                              const float* __restrict__ bc,
                                              float* __restrict__ out) {
  __shared__ float sout[T_ * H_];
  __shared__ float sa1[T_ * D_];
  __shared__ float satt[T_];
  __shared__ float sp[T_];
  __shared__ float sM[H_];
  const int b = blockIdx.x;
  const int tid = threadIdx.x;
  for (int i = tid; i < T_ * H_; i += 128) {
    int t = i / H_, j = i - t * H_;
    sout[i] = hs[(size_t)(t * 32 + b) * H_ + j];
  }
  __syncthreads();
  for (int p = tid; p < T_ * D_; p += 128) {
    int t = p >> 6, d = p & 63;
    float s = ba1[d];
    for (int h = 0; h < H_; ++h) s += sout[t * H_ + h] * wa1[d * H_ + h];
    sa1[t * D_ + d] = tanhf(s);
  }
  __syncthreads();
  if (tid < T_) {
    float s = ba2[0];
    for (int d = 0; d < D_; ++d) s += sa1[tid * D_ + d] * wa2[d];
    satt[tid] = s;
    out[2 * B_ + b * T_ + tid] = s;          // Att (pre-softmax), after Y_prob
  }
  __syncthreads();
  if (tid == 0) {
    float mx = satt[0];
    for (int t = 1; t < T_; ++t) mx = fmaxf(mx, satt[t]);
    float sum = 0.f;
    for (int t = 0; t < T_; ++t) { float e = __expf(satt[t] - mx); sp[t] = e; sum += e; }
    float inv = 1.f / sum;
    for (int t = 0; t < T_; ++t) sp[t] *= inv;
  }
  __syncthreads();
  if (tid < H_) {
    float m = 0.f;
    for (int t = 0; t < T_; ++t) m += sp[t] * sout[t * H_ + tid];
    sM[tid] = m;
  }
  __syncthreads();
  if (tid == 0) {
    float l0 = bc[0], l1 = bc[1];
    for (int h = 0; h < H_; ++h) { l0 += sM[h] * wc[h]; l1 += sM[h] * wc[H_ + h]; }
    float mx = fmaxf(l0, l1);
    float e0 = __expf(l0 - mx), e1 = __expf(l1 - mx);
    float inv = 1.f / (e0 + e1);
    out[b * 2 + 0] = e0 * inv;
    out[b * 2 + 1] = e1 * inv;
  }
}

// ---------------- host launcher ----------------
extern "C" void kernel_launch(void* const* d_in, const int* in_sizes, int n_in,
                              void* d_out, int out_size, void* d_ws, size_t ws_size,
                              hipStream_t stream) {
  (void)in_sizes; (void)n_in; (void)out_size; (void)ws_size;
  const float* x   = (const float*)d_in[0];
  const float* wg  = (const float*)d_in[1];
  const float* bg  = (const float*)d_in[2];
  const float* Wih = (const float*)d_in[3];
  const float* Whh = (const float*)d_in[4];
  const float* bih = (const float*)d_in[5];
  const float* bhh = (const float*)d_in[6];
  const float* wa1 = (const float*)d_in[7];
  const float* ba1 = (const float*)d_in[8];
  const float* wa2 = (const float*)d_in[9];
  const float* ba2 = (const float*)d_in[10];
  const float* wc  = (const float*)d_in[11];
  const float* bc  = (const float*)d_in[12];
  float* out = (float*)d_out;

  char* ws = (char*)d_ws;
  size_t off = 0;
  auto carve = [&](size_t bytes) -> void* {
    void* p = ws + off;
    off = (off + bytes + 255) & ~(size_t)255;
    return p;
  };
  __bf16* xpad = (__bf16*)carve((size_t)RB * KPADX * sizeof(__bf16));   // 1.18 MB
  __bf16* wihb = (__bf16*)carve((size_t)G4 * KPADX * sizeof(__bf16));   // 0.31 MB
  __bf16* whhb = (__bf16*)carve((size_t)G4 * KPADH * sizeof(__bf16));   // 0.10 MB
  float*  gx   = (float*) carve((size_t)RB * G4 * sizeof(float));       // 2.46 MB
  float*  hs   = (float*) carve((size_t)RB * H_ * sizeof(float));       // 0.61 MB

  // 1) weight pack + x K-padding zero-fill
  k_pack<<<600, 256, 0, stream>>>(Wih, Whh, wihb, whhb, xpad);
  // 2) GCN streaming reduction: 1536*360 outputs, one wave each
  k_gcn<<<(RB * N_) / 8, 256, 0, stream>>>(x, wg, bg, xpad);
  // 3) parallel input GEMM (WMMA): 2400 tiles, 8 waves/block
  k_gemm_x<<<300, 256, 0, stream>>>(xpad, wihb, gx);
  // 4) sequential LSTM (WMMA recurrent part, single workgroup)
  k_lstm<<<1, 512, 0, stream>>>(gx, whhb, bih, bhh, hs);
  // 5) attention + classifier, one block per batch element
  k_attn<<<B_, 128, 0, stream>>>(hs, wa1, ba1, wa2, ba2, wc, bc, out);
}
